// Generator_12635793785518
// MI455X (gfx1250) — matrix operand order repaired and checked by reading, not compiled, not run on hardware
//
#include <hip/hip_runtime.h>
#include <hip/hip_bf16.h>

// ---- problem constants (match reference) ----
#define BB 32
#define TT 96
#define INF_ 256
#define HH 512
#define HEADS_ 4
#define VV 32000
#define TP1 97          // T+1 history slots
#define NEGBIG (-1e9f)

typedef _Float16 f16;
typedef __attribute__((ext_vector_type(16))) _Float16 v16h;
typedef __attribute__((ext_vector_type(8)))  float    v8f;

// ======================================================================
// Packing helpers: ISA 7.12.2 16-bit WMMA layouts (wave32)
//  A (16x32, MxK): lane L -> row M = L%16.
//    lane<16 : halves 0..7 -> K=0..7,  halves 8..15 -> K=16..23
//    lane>=16: halves 0..7 -> K=8..15, halves 8..15 -> K=24..31
//  B (32x16, KxN): lane L -> col N = L%16.
//    same K mapping as A (halves 0..7 -> K=0..7/8..15, 8..15 -> K=16..23/24..31)
// Packed storage: [tile][kstep][lane][16 halves] (contiguous v16h per lane)
// ======================================================================
__device__ __forceinline__ int a_klocal(int lane, int h) {
    return (h < 8 ? h : h + 8) + ((lane < 16) ? 0 : 8);
}

__global__ __launch_bounds__(32)
void pack_Bmat(const float* __restrict__ W, f16* __restrict__ out,
               int Ksteps, int ldW, int koff)
{
    int ks = blockIdx.x % Ksteps;
    int nt = blockIdx.x / Ksteps;
    int lane = threadIdx.x;
    int n = nt * 16 + (lane & 15);
    int kb = ks * 32 + ((lane < 16) ? 0 : 8);
    f16* o = out + ((size_t)(nt * Ksteps + ks) * 32 + lane) * 16;
    const float* w = W + (size_t)n * ldW + koff + kb;
#pragma unroll
    for (int h = 0; h < 16; ++h) o[h] = (f16)w[h < 8 ? h : h + 8];
}

__global__ __launch_bounds__(32)
void pack_Bcat(const float* __restrict__ W1, int K1, int ld1,
               const float* __restrict__ W2, int ld2,
               f16* __restrict__ out, int Ksteps)
{
    int ks = blockIdx.x % Ksteps;
    int nt = blockIdx.x / Ksteps;
    int lane = threadIdx.x;
    int n = nt * 16 + (lane & 15);
    int kb = ks * 32 + ((lane < 16) ? 0 : 8);
    f16* o = out + ((size_t)(nt * Ksteps + ks) * 32 + lane) * 16;
#pragma unroll
    for (int h = 0; h < 16; ++h) {
        int k = kb + (h < 8 ? h : h + 8);
        float v = (k < K1) ? W1[(size_t)n * ld1 + k]
                           : W2[(size_t)n * ld2 + (k - K1)];
        o[h] = (f16)v;
    }
}

__global__ __launch_bounds__(32)
void pack_Arow(const float* __restrict__ X, f16* __restrict__ out,
               int Ksteps, int ldX)
{
    int ks = blockIdx.x % Ksteps;
    int mt = blockIdx.x / Ksteps;
    int lane = threadIdx.x;
    int m = mt * 16 + (lane & 15);
    f16* o = out + ((size_t)(mt * Ksteps + ks) * 32 + lane) * 16;
    const float* x = X + (size_t)m * ldX + ks * 32;
#pragma unroll
    for (int h = 0; h < 16; ++h) o[h] = (f16)x[a_klocal(lane, h)];
}

__global__ __launch_bounds__(32)
void pack_Acat2(const float* __restrict__ X1, int K1, int ld1,
                const float* __restrict__ X2, int ld2,
                f16* __restrict__ out, int Ksteps)
{
    int ks = blockIdx.x % Ksteps;
    int mt = blockIdx.x / Ksteps;
    int lane = threadIdx.x;
    int m = mt * 16 + (lane & 15);
    f16* o = out + ((size_t)(mt * Ksteps + ks) * 32 + lane) * 16;
#pragma unroll
    for (int h = 0; h < 16; ++h) {
        int k = ks * 32 + a_klocal(lane, h);
        float v = (k < K1) ? X1[(size_t)m * ld1 + k]
                           : X2[(size_t)m * ld2 + (k - K1)];
        o[h] = (f16)v;
    }
}

__global__ __launch_bounds__(32)
void pack_Acat3(const float* __restrict__ X1, int K1, int ld1,
                const float* __restrict__ X2, int K2, int ld2,
                const float* __restrict__ X3, int ld3,
                f16* __restrict__ out, int Ksteps)
{
    int ks = blockIdx.x % Ksteps;
    int mt = blockIdx.x / Ksteps;
    int lane = threadIdx.x;
    int m = mt * 16 + (lane & 15);
    f16* o = out + ((size_t)(mt * Ksteps + ks) * 32 + lane) * 16;
#pragma unroll
    for (int h = 0; h < 16; ++h) {
        int k = ks * 32 + a_klocal(lane, h);
        float v;
        if (k < K1)            v = X1[(size_t)m * ld1 + k];
        else if (k < K1 + K2)  v = X2[(size_t)m * ld2 + (k - K1)];
        else                   v = X3[(size_t)m * ld3 + (k - K1 - K2)];
        o[h] = (f16)v;
    }
}

// ======================================================================
// Blocked WMMA GEMM: each wave computes 1 M-tile x NB N-tiles.
// A operand staged through LDS via double-buffered async copies with
// STATIC buffer offsets: k-loop manually unrolled by 2 (all Ksteps even)
// and the final pair peeled, so the steady-state body is branch-free:
//   s_wait_asynccnt 2 -> ds_load -> NB WMMAs -> issue next tile (x2).
// 2 async b128 ops per tile, 2 tiles in flight; in-order completion makes
// "asynccnt<=2" retire exactly the oldest tile.
// B streams from L2 via clause'd global_load_b128 + unconditional prefetch.
// ======================================================================
template<int NB>
__global__ __launch_bounds__(32)
void gemm_wmma_nb(const f16* __restrict__ Ap, const f16* __restrict__ Bp,
                  const float* __restrict__ bias, float* __restrict__ D,
                  int Mtiles, int Ksteps, int ldD)
{
    int mt = blockIdx.x % Mtiles;
    int ng = blockIdx.x / Mtiles;
    int lane = threadIdx.x;

    const v16h* Bv[NB];
#pragma unroll
    for (int j = 0; j < NB; ++j)
        Bv[j] = (const v16h*)Bp + (size_t)(ng * NB + j) * Ksteps * 32 + lane;
    v8f acc[NB] = {};

    const f16* Abase = Ap + (size_t)mt * Ksteps * 512;   // 512 halves per 16x32 tile

    __shared__ __align__(32) f16 Asm[2][512];            // double-buffered A tile

    // issue one 16x32 A tile (1 KB) into LDS buffer bufi: 2 x b128 per lane
    auto issueA = [&](int ks, int bufi) {
        unsigned long long ga =
            (unsigned long long)(size_t)(Abase + (size_t)ks * 512 + lane * 16);
        unsigned int lo = (unsigned int)(size_t)(void*)&Asm[bufi][lane * 16];
        asm volatile("global_load_async_to_lds_b128 %0, %1, off"
                     :: "v"(lo), "v"(ga) : "memory");
        asm volatile("global_load_async_to_lds_b128 %0, %1, off offset:16"
                     :: "v"(lo), "v"(ga) : "memory");
    };
    auto fma_tiles = [&](int ks, int bufi) {
        v16h a = *(const v16h*)&Asm[bufi][lane * 16];
#pragma unroll
        for (int j = 0; j < NB; ++j) {
            v16h b = Bv[j][(size_t)ks * 32];
            acc[j] = __builtin_amdgcn_wmma_f32_16x16x32_f16(
                         false, a, false, b, (short)0, acc[j], false, false);
        }
    };

    // prime both LDS buffers (tiles 0 and 1)
    issueA(0, 0);
    issueA(1, 1);

    int ks = 0;
    for (; ks + 2 < Ksteps; ks += 2) {
        asm volatile("s_wait_asynccnt 0x2" ::: "memory");   // tile ks (buf0) ready
#pragma unroll
        for (int j = 0; j < NB; ++j)
            __builtin_prefetch((const void*)(Bv[j] + (size_t)(ks + 8) * 32), 0, 1);
        fma_tiles(ks, 0);
        issueA(ks + 2, 0);                                  // refill buf0
        asm volatile("s_wait_asynccnt 0x2" ::: "memory");   // tile ks+1 (buf1) ready
        fma_tiles(ks + 1, 1);
        issueA(ks + 3, 1);                                  // refill buf1
    }
    // tail: tiles Ksteps-2 (buf0) and Ksteps-1 (buf1) still in flight
    asm volatile("s_wait_asynccnt 0x2" ::: "memory");
    fma_tiles(ks, 0);
    asm volatile("s_wait_asynccnt 0x0" ::: "memory");
    fma_tiles(ks + 1, 1);

    int nl = lane & 15;
    int mh = (lane >> 4) << 3;            // lanes 16..31 hold rows M+8..M+15
#pragma unroll
    for (int j = 0; j < NB; ++j) {
        int n = (ng * NB + j) * 16 + nl;
        float bv = bias ? bias[n] : 0.0f;
#pragma unroll
        for (int r = 0; r < 8; ++r) {
            int m = mt * 16 + mh + r;
            D[(size_t)m * ldD + n] = acc[j][r] + bv;
        }
    }
}

// ======================================================================
// Attention scores: s[b,j,hd] = relu( relu(hpart[b]+bufP[b,j]) . Wa2[hd] + ba2 )
// masked to NEG for j > t. One 64-thread block per (b,j).
// ======================================================================
__global__ __launch_bounds__(64)
void attn_score(const float* __restrict__ hpart, const float* __restrict__ bufP,
                const float* __restrict__ Wa2, const float* __restrict__ ba2,
                float* __restrict__ s, int t)
{
    int b = blockIdx.x / TP1;
    int j = blockIdx.x % TP1;
    int tid = threadIdx.x;
    __shared__ float red[HEADS_][64];
    const float* hp = hpart + (size_t)b * HH;
    const float* bp = bufP + ((size_t)b * TP1 + j) * HH;
    float acc[HEADS_] = {0.f, 0.f, 0.f, 0.f};
    for (int k = tid; k < HH; k += 64) {
        float v = hp[k] + bp[k];
        v = v > 0.f ? v : 0.f;
#pragma unroll
        for (int hd = 0; hd < HEADS_; ++hd) acc[hd] += v * Wa2[hd * HH + k];
    }
#pragma unroll
    for (int hd = 0; hd < HEADS_; ++hd) red[hd][tid] = acc[hd];
    __syncthreads();
    for (int off = 32; off > 0; off >>= 1) {
        if (tid < off)
#pragma unroll
            for (int hd = 0; hd < HEADS_; ++hd) red[hd][tid] += red[hd][tid + off];
        __syncthreads();
    }
    if (tid < HEADS_) {
        float v = red[tid][0] + ba2[tid];
        v = v > 0.f ? v : 0.f;
        if (j > t) v = NEGBIG;
        s[((size_t)b * TP1 + j) * HEADS_ + tid] = v;
    }
}

// ======================================================================
// Softmax over history + context vector. One 128-thread block per (b,head).
// ======================================================================
__global__ __launch_bounds__(128)
void softmax_cvec(const float* __restrict__ s, const float* __restrict__ buf,
                  float* __restrict__ cvec)
{
    int b  = blockIdx.x / HEADS_;
    int hd = blockIdx.x % HEADS_;
    int tid = threadIdx.x;
    __shared__ float w[TP1];
    __shared__ float red[128];
    float v = (tid < TP1) ? s[((size_t)b * TP1 + tid) * HEADS_ + hd] : NEGBIG;
    red[tid] = v; __syncthreads();
    for (int off = 64; off > 0; off >>= 1) {
        if (tid < off) red[tid] = fmaxf(red[tid], red[tid + off]);
        __syncthreads();
    }
    float mx = red[0]; __syncthreads();
    float e = (tid < TP1) ? __expf(v - mx) : 0.f;
    red[tid] = e; __syncthreads();
    for (int off = 64; off > 0; off >>= 1) {
        if (tid < off) red[tid] += red[tid + off];
        __syncthreads();
    }
    float inv = 1.0f / red[0];
    if (tid < TP1) w[tid] = e * inv;
    __syncthreads();
    for (int h = tid; h < HH; h += 128) {
        float a = 0.f;
        for (int j = 0; j < TP1; ++j)
            a += w[j] * buf[((size_t)b * TP1 + j) * HH + h];
        cvec[(size_t)b * (HEADS_ * HH) + hd * HH + h] = a;
    }
}

// ======================================================================
// LSTM cell pointwise (torch gate order i,f,g,o).
// ======================================================================
__global__ __launch_bounds__(256)
void lstm_cell(const float* __restrict__ gates,
               const float* __restrict__ bih, const float* __restrict__ bhh,
               float* __restrict__ h, float* __restrict__ c,
               float* __restrict__ bufrow /* buf + (t+1)*H, stride TP1*H; or null */)
{
    int idx = blockIdx.x * 256 + threadIdx.x;   // 0 .. B*H
    int b = idx >> 9, jj = idx & (HH - 1);
    const float* g = gates + (size_t)b * (4 * HH);
    float ig = g[jj]            + bih[jj]            + bhh[jj];
    float fg = g[jj + HH]       + bih[jj + HH]       + bhh[jj + HH];
    float gg = g[jj + 2 * HH]   + bih[jj + 2 * HH]   + bhh[jj + 2 * HH];
    float og = g[jj + 3 * HH]   + bih[jj + 3 * HH]   + bhh[jj + 3 * HH];
    float si = 1.f / (1.f + __expf(-ig));
    float sf = 1.f / (1.f + __expf(-fg));
    float so = 1.f / (1.f + __expf(-og));
    float c2 = sf * c[idx] + si * tanhf(gg);
    float hn = so * tanhf(c2);
    c[idx] = c2;
    h[idx] = hn;
    if (bufrow) bufrow[(size_t)b * (TP1 * HH) + jj] = hn;
}

// ======================================================================
// Log-softmax over V.
// ======================================================================
__global__ __launch_bounds__(256)
void lsm_reduce(const float* __restrict__ logits, float2* __restrict__ red)
{
    int b = blockIdx.x;
    int tid = threadIdx.x;
    __shared__ float sm[256];
    const float* L = logits + (size_t)b * VV;
    float mx = -1e30f;
    for (int v = tid; v < VV; v += 256) mx = fmaxf(mx, L[v]);
    sm[tid] = mx; __syncthreads();
    for (int off = 128; off > 0; off >>= 1) {
        if (tid < off) sm[tid] = fmaxf(sm[tid], sm[tid + off]);
        __syncthreads();
    }
    mx = sm[0]; __syncthreads();
    float s = 0.f;
    for (int v = tid; v < VV; v += 256) s += __expf(L[v] - mx);
    sm[tid] = s; __syncthreads();
    for (int off = 128; off > 0; off >>= 1) {
        if (tid < off) sm[tid] += sm[tid + off];
        __syncthreads();
    }
    if (tid == 0) red[b] = make_float2(mx, __logf(sm[0]));
}

__global__ __launch_bounds__(256)
void lsm_write(const float* __restrict__ logits, const float2* __restrict__ red,
               float* __restrict__ out /* d_out + t*V, batch stride T*V */)
{
    int idx = blockIdx.x * 256 + threadIdx.x;   // B*V
    int b = idx / VV, v = idx - b * VV;
    float2 r = red[b];
    out[(size_t)b * (TT * VV) + v] = logits[idx] - r.x - r.y;
}

// ======================================================================
// Host driver
// ======================================================================
extern "C" void kernel_launch(void* const* d_in, const int* in_sizes, int n_in,
                              void* d_out, int out_size, void* d_ws, size_t ws_size,
                              hipStream_t stream)
{
    const float* inputs = (const float*)d_in[0];
    const float* W_ih0  = (const float*)d_in[1];
    const float* W_hh0  = (const float*)d_in[2];
    const float* b_ih0  = (const float*)d_in[3];
    const float* b_hh0  = (const float*)d_in[4];
    const float* W_ih1  = (const float*)d_in[5];
    const float* W_hh1  = (const float*)d_in[6];
    const float* b_ih1  = (const float*)d_in[7];
    const float* b_hh1  = (const float*)d_in[8];
    const float* Wa1    = (const float*)d_in[9];
    const float* ba1    = (const float*)d_in[10];
    const float* Wa2    = (const float*)d_in[11];
    const float* ba2    = (const float*)d_in[12];
    const float* Wo     = (const float*)d_in[13];
    const float* bo     = (const float*)d_in[14];
    float* out = (float*)d_out;

    const int LIN0 = INF_ + HEADS_ * HH + HH;   // 2816 = x_t + cvec + h0
    const int KS0  = LIN0 / 32;                 // 88
    const int KS1  = (2 * HH) / 32;             // 32
    const int KOUT = (1 + HEADS_) * HH;         // 2560
    const int KSO  = KOUT / 32;                 // 80
    const int KSH  = HH / 32;                   // 16
    const int NT_G = (4 * HH) / 16;             // 128 gate tiles
    const int NT_H = HH / 16;                   // 32
    const int NT_V = VV / 16;                   // 2000

    // ---- workspace bump allocator (256B aligned) ----
    char* wsb = (char*)d_ws;
    size_t off = 0;
    auto alloc = [&](size_t bytes) -> void* {
        off = (off + 255) & ~(size_t)255;
        void* p = wsb + off;
        off += bytes;
        return p;
    };
    f16* woP   = (f16*)alloc((size_t)NT_V * KSO * 32 * 16 * 2);  // 164 MB, L2-resident
    f16* w0P   = (f16*)alloc((size_t)NT_G * KS0 * 32 * 16 * 2);
    f16* w1P   = (f16*)alloc((size_t)NT_G * KS1 * 32 * 16 * 2);
    f16* wa1aP = (f16*)alloc((size_t)NT_H * KSH * 32 * 16 * 2);
    f16* wa1bP = (f16*)alloc((size_t)NT_H * KSH * 32 * 16 * 2);
    float* buf    = (float*)alloc((size_t)BB * TP1 * HH * 4);
    float* bufP   = (float*)alloc((size_t)BB * TP1 * HH * 4);
    float* h0     = (float*)alloc((size_t)BB * HH * 4);
    float* c0     = (float*)alloc((size_t)BB * HH * 4);
    float* h1     = (float*)alloc((size_t)BB * HH * 4);
    float* c1     = (float*)alloc((size_t)BB * HH * 4);
    float* hpart  = (float*)alloc((size_t)BB * HH * 4);
    float* cvec   = (float*)alloc((size_t)BB * HEADS_ * HH * 4);
    float* scores = (float*)alloc((size_t)BB * TP1 * HEADS_ * 4);
    float* gates  = (float*)alloc((size_t)BB * 4 * HH * 4);
    float* logits = (float*)alloc((size_t)BB * VV * 4);
    float2* red   = (float2*)alloc((size_t)BB * sizeof(float2));
    f16* A_l0  = (f16*)alloc((size_t)2 * KS0 * 32 * 16 * 2);
    f16* A_l1  = (f16*)alloc((size_t)2 * KS1 * 32 * 16 * 2);
    f16* A_h1  = (f16*)alloc((size_t)2 * KSH * 32 * 16 * 2);
    f16* A_out = (f16*)alloc((size_t)2 * KSO * 32 * 16 * 2);

    // ---- one-time weight packing (f32 -> f16 WMMA-B tiles) ----
    pack_Bmat<<<NT_V * KSO, 32, 0, stream>>>(Wo, woP, KSO, KOUT, 0);
    pack_Bcat<<<NT_G * KS0, 32, 0, stream>>>(W_ih0, INF_ + HEADS_ * HH, INF_ + HEADS_ * HH,
                                             W_hh0, HH, w0P, KS0);
    pack_Bcat<<<NT_G * KS1, 32, 0, stream>>>(W_ih1, HH, HH, W_hh1, HH, w1P, KS1);
    pack_Bmat<<<NT_H * KSH, 32, 0, stream>>>(Wa1, wa1aP, KSH, 2 * HH, 0);   // cols 0..511
    pack_Bmat<<<NT_H * KSH, 32, 0, stream>>>(Wa1, wa1bP, KSH, 2 * HH, HH);  // cols 512..1023

    // ---- zero the recurrent state ----
    hipMemsetAsync(buf,  0, (size_t)BB * TP1 * HH * 4, stream);
    hipMemsetAsync(bufP, 0, (size_t)BB * TP1 * HH * 4, stream);
    hipMemsetAsync(h0,   0, (size_t)BB * HH * 4, stream);
    hipMemsetAsync(c0,   0, (size_t)BB * HH * 4, stream);
    hipMemsetAsync(h1,   0, (size_t)BB * HH * 4, stream);
    hipMemsetAsync(c1,   0, (size_t)BB * HH * 4, stream);
    hipMemsetAsync(A_h1, 0, (size_t)2 * KSH * 32 * 16 * 2, stream);

    // ---- sequential time loop ----
    for (int t = 0; t < TT; ++t) {
        // 1) hpart = h1 @ Wa1[:, :H]^T + ba1   (A_h1 packed at end of prev step)
        gemm_wmma_nb<4><<<2 * (NT_H / 4), 32, 0, stream>>>(A_h1, wa1aP, ba1, hpart,
                                                           2, KSH, HH);
        // 2) masked attention scores over history
        attn_score<<<BB * TP1, 64, 0, stream>>>(hpart, bufP, Wa2, ba2, scores, t);
        // 3) softmax + context vectors
        softmax_cvec<<<BB * HEADS_, 128, 0, stream>>>(scores, buf, cvec);
        // 4) layer-0 input pack: [x_t | cvec | h0]
        pack_Acat3<<<2 * KS0, 32, 0, stream>>>(inputs + (size_t)t * INF_, INF_, TT * INF_,
                                               cvec, HEADS_ * HH, HEADS_ * HH,
                                               h0, HH, A_l0, KS0);
        // 5) layer-0 gates GEMM (K = 2816)
        gemm_wmma_nb<4><<<2 * (NT_G / 4), 32, 0, stream>>>(A_l0, w0P, nullptr, gates,
                                                           2, KS0, 4 * HH);
        // 6) cell 0
        lstm_cell<<<(BB * HH) / 256, 256, 0, stream>>>(gates, b_ih0, b_hh0, h0, c0, nullptr);
        // 7) layer-1 input pack: [h0n | h1]
        pack_Acat2<<<2 * KS1, 32, 0, stream>>>(h0, HH, HH, h1, HH, A_l1, KS1);
        // 8) layer-1 gates GEMM (K = 1024)
        gemm_wmma_nb<4><<<2 * (NT_G / 4), 32, 0, stream>>>(A_l1, w1P, nullptr, gates,
                                                           2, KS1, 4 * HH);
        // 9) cell 1 (+ write history slot t+1)
        lstm_cell<<<(BB * HH) / 256, 256, 0, stream>>>(gates, b_ih1, b_hh1, h1, c1,
                                                       buf + (size_t)(t + 1) * HH);
        // 10) pack new h1 (reused for bufP row update now + hpart next step)
        pack_Arow<<<2 * KSH, 32, 0, stream>>>(h1, A_h1, KSH, HH);
        // 11) incremental bufP row: bufP[:, t+1, :] = h1n @ Wa1[:, H:]^T
        gemm_wmma_nb<4><<<2 * (NT_H / 4), 32, 0, stream>>>(A_h1, wa1bP, nullptr,
                                                           bufP + (size_t)(t + 1) * HH,
                                                           2, KSH, TP1 * HH);
        // 12) output-head input pack: [h1n | cvec]
        pack_Acat2<<<2 * KSO, 32, 0, stream>>>(h1, HH, HH, cvec, HEADS_ * HH, A_out, KSO);
        // 13) output head: 32 x 2560 x 32000 f16 WMMA, Wo f16 stays L2-resident
        gemm_wmma_nb<4><<<2 * (NT_V / 4), 32, 0, stream>>>(A_out, woP, bo, logits,
                                                           2, KSO, VV);
        // 14/15) log-softmax and write out[:, t, :]
        lsm_reduce<<<BB, 256, 0, stream>>>(logits, red);
        lsm_write<<<(BB * VV) / 256, 256, 0, stream>>>(logits, red, out + (size_t)t * VV);
    }
}
